// KPNextBlock_16157666968110
// MI455X (gfx1250) — matrix operand order repaired and checked by reading, not compiled, not run on hardware
//
#include <hip/hip_runtime.h>
#include <math.h>

typedef __attribute__((ext_vector_type(16))) _Float16 v16h;
typedef __attribute__((ext_vector_type(8)))  _Float16 v8h;
typedef __attribute__((ext_vector_type(8)))  float    v8f;

#define M_PTS    50000
#define H_NB     32
#define C_FEAT   128
#define K_KP     15
#define CPG      16
#define SIGMA_INV (1.0f/0.9f)

#define NTILE1   8      // 128/16 N-tiles, GEMM1
#define NTILE2   15     // 240/16 N-tiles, GEMM2
#define KCH      4      // 128/32 K-chunks
#define MTILES   5      // row-tiles per wave: 3125 tiles = 625 blocks * 5 (exact)
#define HS       136    // LDS h-tile row stride in halves (bank-skewed)

#define W1H_ELEMS (NTILE1*KCH*32*16)   // 16384 halves
#define W2H_ELEMS (NTILE2*KCH*32*16)   // 30720 halves

// ---------------------------------------------------------------------------
// Kernel 0: pre-swizzle w1/w2 (f32, row-major [K][N]) into the CDNA5 WMMA
// B-operand layout (f16).  B element e of lane l:  K = (e<8 ? e : e+8) + 8*(l>>4),
// N = l&15  (per 32-K chunk / 16-N tile).  Stored so each lane's 16 halves
// are contiguous: idx = ((nt*KCH + kc)*32 + lane)*16 + e.
// ---------------------------------------------------------------------------
__global__ __launch_bounds__(256) void kp_prep_kernel(
    const float* __restrict__ w1, const float* __restrict__ w2,
    _Float16* __restrict__ w1h, _Float16* __restrict__ w2h)
{
  int tid = blockIdx.x * 256 + threadIdx.x;
  if (tid < W1H_ELEMS) {
    int e  = tid & 15;
    int l  = (tid >> 4) & 31;
    int kc = (tid >> 9) & 3;
    int nt = tid >> 11;
    int k  = kc * 32 + (e < 8 ? e : e + 8) + 8 * (l >> 4);
    int n  = nt * 16 + (l & 15);
    w1h[tid] = (_Float16)w1[k * 128 + n];
  } else {
    int idx = tid - W1H_ELEMS;
    if (idx < W2H_ELEMS) {
      int e  = idx & 15;
      int l  = (idx >> 4) & 31;
      int kc = (idx >> 9) & 3;
      int nt = idx >> 11;
      int k  = kc * 32 + (e < 8 ? e : e + 8) + 8 * (l >> 4);
      int n  = nt * 16 + (l & 15);
      w2h[idx] = (_Float16)w2[k * 240 + n];
    }
  }
}

static __device__ inline v16h pack16(float4 f0, float4 f1, float4 f2, float4 f3) {
  v16h r;
  r[0]=(_Float16)f0.x; r[1]=(_Float16)f0.y; r[2]=(_Float16)f0.z; r[3]=(_Float16)f0.w;
  r[4]=(_Float16)f1.x; r[5]=(_Float16)f1.y; r[6]=(_Float16)f1.z; r[7]=(_Float16)f1.w;
  r[8]=(_Float16)f2.x; r[9]=(_Float16)f2.y; r[10]=(_Float16)f2.z; r[11]=(_Float16)f2.w;
  r[12]=(_Float16)f3.x; r[13]=(_Float16)f3.y; r[14]=(_Float16)f3.z; r[15]=(_Float16)f3.w;
  return r;
}

// ---------------------------------------------------------------------------
// Kernel 1: modulation MLP.  One wave per block; each wave owns MTILES=5
// consecutive 16-row tiles so every B-operand load (w1h/w2h, L2-resident)
// is reused 5x -> B traffic drops from ~287MB to ~57MB.
//   GEMM1: h = leaky_relu(x @ w1 + b1)   (bias via WMMA C operand)
//   GEMM2: mod = sigmoid(h @ w2)         (h transposed through LDS into the
//                                         16-bit A layout: K(e,half) =
//                                         (e<8 ? e : e+8) + 8*half per chunk)
// Modulations stored f16 to workspace ([M,240]).
// ---------------------------------------------------------------------------
__global__ __launch_bounds__(32) void kp_mlp_kernel(
    const float* __restrict__ s_feats, const float* __restrict__ b1,
    const _Float16* __restrict__ w1h, const _Float16* __restrict__ w2h,
    _Float16* __restrict__ mods)
{
  __shared__ _Float16 h_sh[MTILES][16 * HS];   // 21.8 KB

  const int lane = threadIdx.x & 31;
  const int half = lane >> 4;
  const int lrow = lane & 15;
  const int tile0 = blockIdx.x * MTILES;       // grid exact: 625*5 = 3125

  // ---- A tiles: 5x (16x128) f32 -> f16, WMMA 16-bit A layout ----
  v16h a[MTILES][KCH];
  #pragma unroll
  for (int t = 0; t < MTILES; ++t) {
    const float* xrow = s_feats + (size_t)((tile0 + t) * 16 + lrow) * C_FEAT;
    #pragma unroll
    for (int kc = 0; kc < KCH; ++kc) {
      const int b0 = kc * 32 + 8 * half;
      float4 f0 = *(const float4*)(xrow + b0);
      float4 f1 = *(const float4*)(xrow + b0 + 4);
      float4 f2 = *(const float4*)(xrow + b0 + 16);
      float4 f3 = *(const float4*)(xrow + b0 + 20);
      a[t][kc] = pack16(f0, f1, f2, f3);
    }
  }

  // ---- GEMM1 + bias + leaky_relu -> h tiles in LDS (f16) ----
  #pragma unroll
  for (int nt = 0; nt < NTILE1; ++nt) {
    const float bv = b1[nt * 16 + lrow];
    v16h bt[KCH];
    #pragma unroll
    for (int kc = 0; kc < KCH; ++kc)
      bt[kc] = *(const v16h*)(w1h + ((size_t)(nt * KCH + kc) * 32 + lane) * 16);
    #pragma unroll
    for (int t = 0; t < MTILES; ++t) {
      v8f acc = {bv, bv, bv, bv, bv, bv, bv, bv};
      #pragma unroll
      for (int kc = 0; kc < KCH; ++kc)
        acc = __builtin_amdgcn_wmma_f32_16x16x32_f16(
            false, a[t][kc], false, bt[kc], (short)0, acc, false, false);
      #pragma unroll
      for (int r = 0; r < 8; ++r) {
        float v = acc[r];
        v = (v >= 0.0f) ? v : 0.1f * v;          // leaky_relu 0.1
        h_sh[t][(r + 8 * half) * HS + nt * 16 + lrow] = (_Float16)v;
      }
    }
  }

  __syncthreads();   // single-wave WG: barrier is a NOP, fence orders LDS

  // ---- reload h tiles in A layout (overwrites a[][]) ----
  #pragma unroll
  for (int t = 0; t < MTILES; ++t) {
    const _Float16* hb = &h_sh[t][lrow * HS];
    #pragma unroll
    for (int kc = 0; kc < KCH; ++kc) {
      const int b0 = kc * 32 + 8 * half;
      v8h lo = *(const v8h*)(hb + b0);
      v8h hi = *(const v8h*)(hb + b0 + 16);
      v16h tt;
      #pragma unroll
      for (int e = 0; e < 8; ++e) { tt[e] = lo[e]; tt[e + 8] = hi[e]; }
      a[t][kc] = tt;
    }
  }

  // ---- GEMM2 + sigmoid -> modulations (f16) ----
  #pragma unroll
  for (int nt = 0; nt < NTILE2; ++nt) {
    v16h bt[KCH];
    #pragma unroll
    for (int kc = 0; kc < KCH; ++kc)
      bt[kc] = *(const v16h*)(w2h + ((size_t)(nt * KCH + kc) * 32 + lane) * 16);
    #pragma unroll
    for (int t = 0; t < MTILES; ++t) {
      v8f acc = {};
      #pragma unroll
      for (int kc = 0; kc < KCH; ++kc)
        acc = __builtin_amdgcn_wmma_f32_16x16x32_f16(
            false, a[t][kc], false, bt[kc], (short)0, acc, false, false);
      const size_t m0 = (size_t)(tile0 + t) * 16;
      #pragma unroll
      for (int r = 0; r < 8; ++r) {
        float s = 1.0f / (1.0f + expf(-acc[r]));
        mods[(m0 + r + 8 * half) * 240 + nt * 16 + lrow] = (_Float16)s;
      }
    }
  }
}

// ---------------------------------------------------------------------------
// Kernel 2: geometry + gather-aggregate.  One wave per output row.
// Phase 1 (lane == neighbor h): nn_idx / influence.  Phase 2 (lane owns 4
// channels): out[m,c] = sum_h feats[ind,c]*weights[nn,c]*mod[m,nn*16+c/8]*infl.
// Zero-influence neighbors (common for random far neighbors) are skipped with
// a wave-uniform branch, eliminating their feature gather entirely.
// ---------------------------------------------------------------------------
__global__ __launch_bounds__(256) void kp_agg_kernel(
    const float* __restrict__ q_pts, const float* __restrict__ s_pts,
    const float* __restrict__ s_feats, const int* __restrict__ neighb,
    const float* __restrict__ da_scale, const float* __restrict__ weights,
    const float* __restrict__ kpts, const _Float16* __restrict__ mods,
    float* __restrict__ out)
{
  __shared__ float sw[K_KP * C_FEAT];       // 15x128 shared conv weights
  __shared__ float mod_sh[8][240];
  __shared__ int   ind_sh[8][H_NB];
  __shared__ int   nn_sh[8][H_NB];
  __shared__ float infl_sh[8][H_NB];

  const int t = threadIdx.x;
  for (int j = t; j < K_KP * C_FEAT; j += 256) sw[j] = weights[j];

  const int wv   = t >> 5;
  const int lane = t & 31;
  const int m    = blockIdx.x * 8 + wv;     // grid exact: 6250*8 = 50000

  // ---- phase 1: per-lane neighbor (h == lane) ----
  const int   ind = neighb[m * H_NB + lane];
  const float qx = q_pts[m * 3 + 0], qy = q_pts[m * 3 + 1], qz = q_pts[m * 3 + 2];
  const float da = da_scale[m];
  const float nx = s_pts[(size_t)ind * 3 + 0] - qx;
  const float ny = s_pts[(size_t)ind * 3 + 1] - qy;
  const float nz = s_pts[(size_t)ind * 3 + 2] - qz;

  float best = 1.0e30f; int bi = 0;
  #pragma unroll
  for (int k = 0; k < K_KP; ++k) {
    float dx = nx - da * kpts[k * 3 + 0];
    float dy = ny - da * kpts[k * 3 + 1];
    float dz = nz - da * kpts[k * 3 + 2];
    float d = dx * dx + dy * dy + dz * dz;
    if (d < best) { best = d; bi = k; }     // strict < -> first argmin, as jnp
  }
  float fl = 1.0f - sqrtf(best) * SIGMA_INV;
  fl = fl > 0.0f ? fl : 0.0f;

  ind_sh[wv][lane]  = ind;
  nn_sh[wv][lane]   = bi;
  infl_sh[wv][lane] = fl;
  for (int j = lane; j < 240; j += 32)
    mod_sh[wv][j] = (float)mods[(size_t)m * 240 + j];

  __syncthreads();

  // ---- phase 2: lane owns channels [4*lane, 4*lane+4) ----
  const int c0 = lane * 4;
  const int mc = lane >> 1;                 // c/8, identical for the 4 channels
  float ax = 0.0f, ay = 0.0f, az = 0.0f, aw = 0.0f;
  #pragma unroll 4
  for (int h = 0; h < H_NB; ++h) {
    float f = infl_sh[wv][h];               // uniform across the wave
    if (f > 0.0f) {
      int ni = ind_sh[wv][h];
      int nn = nn_sh[wv][h];
      float4 ft = *(const float4*)(s_feats + (size_t)ni * C_FEAT + c0);
      float4 wt = *(const float4*)(sw + nn * C_FEAT + c0);
      float  s  = f * mod_sh[wv][nn * CPG + mc];
      ax = fmaf(ft.x, wt.x * s, ax);
      ay = fmaf(ft.y, wt.y * s, ay);
      az = fmaf(ft.z, wt.z * s, az);
      aw = fmaf(ft.w, wt.w * s, aw);
    }
  }
  *(float4*)(out + (size_t)m * C_FEAT + c0) = make_float4(ax, ay, az, aw);
}

// ---------------------------------------------------------------------------
extern "C" void kernel_launch(void* const* d_in, const int* in_sizes, int n_in,
                              void* d_out, int out_size, void* d_ws, size_t ws_size,
                              hipStream_t stream) {
  (void)in_sizes; (void)n_in; (void)out_size; (void)ws_size;
  const float* q_pts   = (const float*)d_in[0];
  const float* s_pts   = (const float*)d_in[1];
  const float* s_feats = (const float*)d_in[2];
  const int*   neighb  = (const int*)  d_in[3];
  const float* da      = (const float*)d_in[4];
  const float* weights = (const float*)d_in[5];
  const float* w1      = (const float*)d_in[6];
  const float* b1      = (const float*)d_in[7];
  const float* w2      = (const float*)d_in[8];
  const float* kpts    = (const float*)d_in[9];
  float* out = (float*)d_out;

  // workspace: [w1h 32KB][w2h 60KB][mods f16 24MB]  (all 32B aligned)
  _Float16* w1h  = (_Float16*)d_ws;
  _Float16* w2h  = w1h + W1H_ELEMS;
  _Float16* mods = w2h + W2H_ELEMS;

  kp_prep_kernel<<<(W1H_ELEMS + W2H_ELEMS) / 256, 256, 0, stream>>>(w1, w2, w1h, w2h);
  kp_mlp_kernel<<<625, 32, 0, stream>>>(s_feats, b1, w1h, w2h, mods);
  kp_agg_kernel<<<6250, 256, 0, stream>>>(q_pts, s_pts, s_feats, neighb, da,
                                          weights, kpts, mods, out);
}
